// RNNDecoder_30288109371801
// MI455X (gfx1250) — matrix operand order repaired and checked by reading, not compiled, not run on hardware
//
#include <hip/hip_runtime.h>
#include <hip/hip_bf16.h>

#define B_   64
#define S_   128
#define H_   1024
#define E_   512
#define V_   50000
#define MAXC 150
#define VOFF_ (V_ + MAXC)

typedef _Float16     v8h   __attribute__((ext_vector_type(8)));
typedef _Float16     v16h  __attribute__((ext_vector_type(16)));
typedef float        v8f   __attribute__((ext_vector_type(8)));
typedef unsigned int u32x4 __attribute__((ext_vector_type(4)));
typedef int          i32x4 __attribute__((ext_vector_type(4)));
typedef int          i32x8 __attribute__((ext_vector_type(8)));

#if defined(__has_builtin)
#if __has_builtin(__builtin_amdgcn_tensor_load_to_lds)
#define HAVE_TDM 1
#endif
#endif

#if HAVE_TDM
// ---------------------------------------------------------------------------
// Issue one TDM 2D tile load: global (row-major, stride_x elements between
// rows, 4-byte elements) -> LDS linear [tile_y][tile_x]. Descriptor packing
// per CDNA5 ISA sec. 8.3 (group0) / 8.4 (group1). Rows y >= tensor_y are
// zero-filled by TDM OOB handling (used for the N-remainder of out_W).
// ---------------------------------------------------------------------------
__device__ __forceinline__ void tdm_load_tile_2d(unsigned lds_addr, const float* gaddr,
                                                 int tile_x, int tile_y,
                                                 long tensor_x, long tensor_y,
                                                 long stride_x) {
    unsigned long long ga = (unsigned long long)(const void*)gaddr;
    u32x4 g0;
    g0[0] = 1u;                                        // count=1 (valid), user mode
    g0[1] = lds_addr;                                  // lds_addr [63:32]
    g0[2] = (unsigned)(ga & 0xffffffffull);            // global_addr [95:64]
    g0[3] = (unsigned)((ga >> 32) & 0x1ffffffull)      // global_addr [120:96]
          | (2u << 30);                                // type=2 ("image")
    i32x8 g1;
    g1[0] = (int)(2u << 16);                           // data_size=2 (4 bytes)
    g1[1] = (int)(((unsigned)tensor_x & 0xffffu) << 16);          // dim0[15:0] @ [63:48]
    g1[2] = (int)((((unsigned)(tensor_x >> 16)) & 0xffffu)
          | (((unsigned)tensor_y & 0xffffu) << 16));              // dim0[31:16] | dim1[15:0]
    g1[3] = (int)((((unsigned)(tensor_y >> 16)) & 0xffffu)
          | (((unsigned)tile_x & 0xffffu) << 16));                // dim1[31:16] | tile_dim0
    g1[4] = (int)((unsigned)tile_y & 0xffffu);                    // tile_dim1 (tile_dim2=0)
    g1[5] = (int)(unsigned)(stride_x & 0xffffffffll);             // dim0_stride[31:0]
    g1[6] = (int)((unsigned)((stride_x >> 32) & 0xffffll));       // dim0_stride[47:32]
    g1[7] = 0;                                                    // dim1_stride=0 (2D)
    i32x4 gz = {0, 0, 0, 0};
#if __clang_major__ >= 23
    i32x8 gz8 = {0, 0, 0, 0, 0, 0, 0, 0};
    __builtin_amdgcn_tensor_load_to_lds(g0, g1, gz, gz, gz8, 0);
#else
    __builtin_amdgcn_tensor_load_to_lds(g0, g1, gz, gz, 0);
#endif
}
#endif // HAVE_TDM

// ---------------------------------------------------------------------------
// WMMA GEMM: C[M=64, N] = A[64, K] * W[N, K]^T (+ bias), fp32 in/out,
// f16 WMMA with f32 accumulation. Block = 256 threads = 8 waves.
// Block tile: 64 (M) x 32 (N); each wave owns one 16x16 C tile.
// Tiles are fed by the Tensor Data Mover into double-buffered LDS (fp32);
// wave 0 issues the next tile pair, s_wait_tensorcnt 2 overlaps DMA w/ WMMA.
// Requires: K % 32 == 0, M == 64 (true for every layer in this model).
// ---------------------------------------------------------------------------
__global__ __launch_bounds__(256)
void wmma_gemm_atb(const float* __restrict__ A, const float* __restrict__ W,
                   const float* __restrict__ bias, float* __restrict__ C,
                   int N, int K) {
    __shared__ __align__(16) float As[2][64 * 32];   // M x K fp32 tiles (2 x 8KB)
    __shared__ __align__(16) float Ws[2][32 * 32];   // N x K fp32 tiles (2 x 4KB)
    const int tid  = threadIdx.x;
    const int wave = tid >> 5;
    const int lane = tid & 31;
    const int lr   = lane & 15;     // row/col within 16
    const int hh   = lane >> 4;     // lane half (0/1)
    const int m0   = (wave & 3) * 16;
    const int n0   = (wave >> 2) * 16;
    const int nblk = blockIdx.x * 32;
    const int steps = K >> 5;       // K / 32
    const long nrem = (long)N - (long)nblk;   // rows of W present (TDM zero-fills rest)

#if HAVE_TDM
    const bool issuer = (wave == 0);
    const unsigned ldsA0 = (unsigned)(unsigned long long)(const void*)&As[0][0];
    const unsigned ldsA1 = (unsigned)(unsigned long long)(const void*)&As[1][0];
    const unsigned ldsW0 = (unsigned)(unsigned long long)(const void*)&Ws[0][0];
    const unsigned ldsW1 = (unsigned)(unsigned long long)(const void*)&Ws[1][0];
    if (issuer) {
        // prime buffer 0 with the k0 = 0 tile pair
        tdm_load_tile_2d(ldsA0, A,                   32, 64, K, 64,   K);
        tdm_load_tile_2d(ldsW0, W + (size_t)nblk * K, 32, 32, K, nrem, K);
    }
#endif

    v8f acc = {};   // 16x16 f32 accumulator fragment

    for (int s = 0; s < steps; ++s) {
        const int cur = s & 1;
#if HAVE_TDM
        if (issuer) {
            if (s + 1 < steps) {
                const int k1 = (s + 1) << 5;
                const unsigned la = cur ? ldsA0 : ldsA1;
                const unsigned lw = cur ? ldsW0 : ldsW1;
                tdm_load_tile_2d(la, A + k1,                        32, 64, K - k1, 64,   K);
                tdm_load_tile_2d(lw, W + (size_t)nblk * K + k1,     32, 32, K - k1, nrem, K);
                // in-order completion: <=2 outstanding => current pair has landed
                __builtin_amdgcn_s_wait_tensorcnt(2);
            } else {
                __builtin_amdgcn_s_wait_tensorcnt(0);
            }
        }
#else
        {   // fallback: cooperative fp32 tile load (no double buffering benefit)
            const int k0 = s << 5;
            #pragma unroll
            for (int i = 0; i < 2; ++i) {
                int f  = tid + i * 256;
                int r  = f >> 3;
                int c4 = (f & 7) << 2;
                *(float4*)&As[cur][r * 32 + c4] =
                    *(const float4*)(A + (size_t)r * K + k0 + c4);
            }
            int r  = tid >> 3;
            int c4 = (tid & 7) << 2;
            int n  = nblk + r;
            float4 v = make_float4(0.f, 0.f, 0.f, 0.f);
            if (n < N) v = *(const float4*)(W + (size_t)n * K + k0 + c4);
            *(float4*)&Ws[cur][r * 32 + c4] = v;
        }
#endif
        __syncthreads();   // tiles in buf[cur] visible to all waves

        // ---- build fragments per ISA 7.12.2 layouts (fp32 LDS -> f16 regs) ----
        // A (16x32 f16): lane half hh -> e=0..7 : K=hh*8+e ; e=8..15 : K=16+hh*8+(e-8)
        const float* Ab = &As[cur][(m0 + lr) * 32];
        float4 fa0 = *(const float4*)(Ab + hh * 8);
        float4 fa1 = *(const float4*)(Ab + hh * 8 + 4);
        float4 fa2 = *(const float4*)(Ab + 16 + hh * 8);
        float4 fa3 = *(const float4*)(Ab + 16 + hh * 8 + 4);
        v16h a;
        a[0]  = (_Float16)fa0.x; a[1]  = (_Float16)fa0.y; a[2]  = (_Float16)fa0.z; a[3]  = (_Float16)fa0.w;
        a[4]  = (_Float16)fa1.x; a[5]  = (_Float16)fa1.y; a[6]  = (_Float16)fa1.z; a[7]  = (_Float16)fa1.w;
        a[8]  = (_Float16)fa2.x; a[9]  = (_Float16)fa2.y; a[10] = (_Float16)fa2.z; a[11] = (_Float16)fa2.w;
        a[12] = (_Float16)fa3.x; a[13] = (_Float16)fa3.y; a[14] = (_Float16)fa3.z; a[15] = (_Float16)fa3.w;
        // B (32x16 f16): column n0+lr, lane half hh -> e=0..7 : K=hh*8+e ; e=8..15 : K=16+hh*8+(e-8)
        const float* Bb = &Ws[cur][(n0 + lr) * 32 + hh * 8];
        float4 fb0 = *(const float4*)(Bb + 0);
        float4 fb1 = *(const float4*)(Bb + 4);
        float4 fb2 = *(const float4*)(Bb + 16);
        float4 fb3 = *(const float4*)(Bb + 20);
        v16h b;
        b[0]  = (_Float16)fb0.x; b[1]  = (_Float16)fb0.y; b[2]  = (_Float16)fb0.z; b[3]  = (_Float16)fb0.w;
        b[4]  = (_Float16)fb1.x; b[5]  = (_Float16)fb1.y; b[6]  = (_Float16)fb1.z; b[7]  = (_Float16)fb1.w;
        b[8]  = (_Float16)fb2.x; b[9]  = (_Float16)fb2.y; b[10] = (_Float16)fb2.z; b[11] = (_Float16)fb2.w;
        b[12] = (_Float16)fb3.x; b[13] = (_Float16)fb3.y; b[14] = (_Float16)fb3.z; b[15] = (_Float16)fb3.w;

        acc = __builtin_amdgcn_wmma_f32_16x16x32_f16(
            /*neg_a=*/false, a, /*neg_b=*/false, b,
            /*c_mod=*/(short)0, acc, /*reuse_a=*/false, /*reuse_b=*/false);
        __syncthreads();   // all reads of buf[cur] done before TDM refills it
    }

    // ---- store: VGPR j -> M = m0 + hh*8 + j, N = nblk + n0 + lr ----
    #pragma unroll
    for (int j = 0; j < 8; ++j) {
        int m = m0 + hh * 8 + j;
        int n = nblk + n0 + lr;
        if (n < N) {
            float v = acc[j];
            if (bias) v += bias[n];
            C[(size_t)m * N + n] = v;
        }
    }
}

// ---------------------------------------------------------------------------
// Copy-aware embedding gather/blend; also writes the embedding half of rnn_in.
// ---------------------------------------------------------------------------
__global__ void embed_blend(const int* __restrict__ tok,
                            const float* __restrict__ table,
                            const float* __restrict__ copy_bank,
                            float* __restrict__ embedded,
                            float* __restrict__ rnn_in) {
    int i = blockIdx.x * blockDim.x + threadIdx.x;
    if (i >= B_ * E_) return;
    int b = i / E_, e = i % E_;
    int t  = tok[b];
    int ci = t - V_;
    float v = (ci >= 0)
        ? copy_bank[(size_t)b * S_ * E_ + (size_t)ci * E_ + e]
        : table[(size_t)t * E_ + e];
    embedded[i] = v;
    rnn_in[(size_t)b * (H_ + E_) + e] = v;
}

// ---------------------------------------------------------------------------
// Masked attention: scores = q . mem, softmax over valid positions, context.
// One block per batch row; wave32 __shfl_xor reductions for the dots.
// ---------------------------------------------------------------------------
__global__ __launch_bounds__(256)
void attn_softmax_ctx(const float* __restrict__ q, const float* __restrict__ mem,
                      const int* __restrict__ len, int min_pos,
                      float* __restrict__ attn_out,      // (B,S) slice of d_out
                      float* __restrict__ context,       // (B,H) or nullptr
                      float* __restrict__ rnn_in) {      // rnn_in[:,E:] or nullptr
    const int b = blockIdx.x;
    __shared__ float sc[S_];
    __shared__ float red[256];
    const int tid = threadIdx.x, wave = tid >> 5, lane = tid & 31;
    const float* qb = q + (size_t)b * H_;
    const float* mb = mem + (size_t)b * S_ * H_;

    for (int s = wave; s < S_; s += 8) {
        const float* ms = mb + (size_t)s * H_;
        float p = 0.f;
        for (int h = lane; h < H_; h += 32) p += qb[h] * ms[h];
        #pragma unroll
        for (int off = 16; off; off >>= 1) p += __shfl_xor(p, off, 32);
        if (lane == 0) sc[s] = p;
    }
    __syncthreads();

    const int L = len[b];
    float m = -3.0e38f;
    if (tid < S_) {
        bool ok = (tid >= min_pos) && (tid < L);
        m = ok ? sc[tid] : -1e9f;
        sc[tid] = m;
    }
    red[tid] = m;
    __syncthreads();
    for (int off = 128; off; off >>= 1) {
        if (tid < off) red[tid] = fmaxf(red[tid], red[tid + off]);
        __syncthreads();
    }
    const float mx = red[0];
    __syncthreads();

    float e = (tid < S_) ? __expf(sc[tid] - mx) : 0.f;
    if (tid < S_) sc[tid] = e;
    red[tid] = e;
    __syncthreads();
    for (int off = 128; off; off >>= 1) {
        if (tid < off) red[tid] += red[tid + off];
        __syncthreads();
    }
    const float inv = 1.f / red[0];
    __syncthreads();

    if (tid < S_) {
        float al = sc[tid] * inv;
        sc[tid] = al;
        attn_out[(size_t)b * S_ + tid] = al;
    }
    __syncthreads();

    if (context != nullptr) {
        for (int h = tid; h < H_; h += 256) {
            float acc = 0.f;
            for (int s = 0; s < S_; ++s) acc += sc[s] * mb[(size_t)s * H_ + h];
            context[(size_t)b * H_ + h] = acc;
            rnn_in[(size_t)b * (H_ + E_) + E_ + h] = acc;
        }
    }
}

// ---------------------------------------------------------------------------
// GRU gates (PyTorch r,z,n convention); biases already in gi/gh from GEMM.
// ---------------------------------------------------------------------------
__global__ void gru_gates(const float* __restrict__ gi, const float* __restrict__ gh,
                          const float* __restrict__ hprev,
                          float* __restrict__ hnew, float* __restrict__ hout2) {
    int i = blockIdx.x * blockDim.x + threadIdx.x;
    if (i >= B_ * H_) return;
    int b = i / H_, h = i % H_;
    const float* gib = gi + (size_t)b * 3 * H_;
    const float* ghb = gh + (size_t)b * 3 * H_;
    float r = 1.f / (1.f + __expf(-(gib[h] + ghb[h])));
    float z = 1.f / (1.f + __expf(-(gib[H_ + h] + ghb[H_ + h])));
    float n = tanhf(gib[2 * H_ + h] + r * ghb[2 * H_ + h]);
    float hv = (1.f - z) * n + z * hprev[i];
    hnew[i]  = hv;
    hout2[i] = hv;
}

// context_output = [h1 ; context ; embedded]  (B, 2H+E)
__global__ void concat_ctxout(const float* __restrict__ h1,
                              const float* __restrict__ context,
                              const float* __restrict__ embedded,
                              float* __restrict__ co) {
    const int CO = 2 * H_ + E_;
    int i = blockIdx.x * blockDim.x + threadIdx.x;
    if (i >= B_ * CO) return;
    int b = i / CO, c = i % CO;
    float v;
    if (c < H_)           v = h1[(size_t)b * H_ + c];
    else if (c < 2 * H_)  v = context[(size_t)b * H_ + (c - H_)];
    else                  v = embedded[(size_t)b * E_ + (c - 2 * H_)];
    co[i] = v;
}

// selector = sigmoid(context_output . copy_sel_w) ; one block per batch row
__global__ __launch_bounds__(256)
void selector_k(const float* __restrict__ co, const float* __restrict__ w,
                float* __restrict__ sel_ws, float* __restrict__ sel_out) {
    const int CO = 2 * H_ + E_;
    int b = blockIdx.x, tid = threadIdx.x;
    __shared__ float red[256];
    float p = 0.f;
    for (int c = tid; c < CO; c += 256) p += co[(size_t)b * CO + c] * w[c];
    red[tid] = p;
    __syncthreads();
    for (int off = 128; off; off >>= 1) {
        if (tid < off) red[tid] += red[tid + off];
        __syncthreads();
    }
    if (tid == 0) {
        float s = 1.f / (1.f + __expf(-red[0]));
        sel_ws[b]  = s;
        sel_out[b] = s;
    }
}

// softmax over V=50000 logits, blend with copy probs, log; one block per row
__global__ __launch_bounds__(256)
void final_blend(const float* __restrict__ logits, const float* __restrict__ sel,
                 const float* __restrict__ copy_w, float* __restrict__ out) {
    int b = blockIdx.x, tid = threadIdx.x;
    __shared__ float red[256];
    const float* lb = logits + (size_t)b * V_;

    float m = -3.0e38f;
    for (int v = tid; v < V_; v += 256) m = fmaxf(m, lb[v]);
    red[tid] = m;
    __syncthreads();
    for (int off = 128; off; off >>= 1) {
        if (tid < off) red[tid] = fmaxf(red[tid], red[tid + off]);
        __syncthreads();
    }
    const float mx = red[0];
    __syncthreads();

    float s = 0.f;
    for (int v = tid; v < V_; v += 256) s += __expf(lb[v] - mx);
    red[tid] = s;
    __syncthreads();
    for (int off = 128; off; off >>= 1) {
        if (tid < off) red[tid] += red[tid + off];
        __syncthreads();
    }
    const float inv = 1.f / red[0];

    const float g = sel[b];
    const float om = 1.f - g;
    float* ob = out + (size_t)b * VOFF_;
    for (int v = tid; v < V_; v += 256)
        ob[v] = __logf(__expf(lb[v] - mx) * inv * om + 1e-20f);
    for (int v = V_ + tid; v < VOFF_; v += 256) {
        int si = v - V_;
        float cp = (si < S_) ? copy_w[(size_t)b * S_ + si] * g : 0.f;
        ob[v] = __logf(cp + 1e-20f);
    }
}

// ---------------------------------------------------------------------------
extern "C" void kernel_launch(void* const* d_in, const int* in_sizes, int n_in,
                              void* d_out, int out_size, void* d_ws, size_t ws_size,
                              hipStream_t stream) {
    const int*   tok      = (const int*)  d_in[0];
    const float* last_h   = (const float*)d_in[1];   // (2,B,H)
    const float* mem      = (const float*)d_in[2];   // (B,S,H)
    const float* copybank = (const float*)d_in[3];   // (B,S,E)
    const int*   enc_len  = (const int*)  d_in[4];
    const float* table    = (const float*)d_in[5];   // (VOFF,E)
    const float* attn_W   = (const float*)d_in[6];   // (H,H)
    const float* cattn_W  = (const float*)d_in[7];   // (H,H)
    const float* sel_w    = (const float*)d_in[8];   // (2H+E,)
    const float* out_W    = (const float*)d_in[9];   // (V,2H+E)
    const float* out_b    = (const float*)d_in[10];  // (V,)
    const float* Wih0     = (const float*)d_in[11];  // (3H, H+E)
    const float* Whh0     = (const float*)d_in[12];  // (3H, H)
    const float* bih0     = (const float*)d_in[13];
    const float* bhh0     = (const float*)d_in[14];
    const float* Wih1     = (const float*)d_in[15];  // (3H, H)
    const float* Whh1     = (const float*)d_in[16];  // (3H, H)
    const float* bih1     = (const float*)d_in[17];
    const float* bhh1     = (const float*)d_in[18];

    // ---- workspace layout (floats) ----
    float* ws = (float*)d_ws;
    float* emb    = ws;                 // 64*512
    float* rnn_in = ws + 32768;         // 64*1536
    float* qm     = ws + 131072;        // 64*1024
    float* qc     = ws + 196608;        // 64*1024
    float* ctx    = ws + 262144;        // 64*1024
    float* gi     = ws + 327680;        // 64*3072
    float* gh     = ws + 524288;        // 64*3072
    float* h0     = ws + 720896;        // 64*1024
    float* h1     = ws + 786432;        // 64*1024
    float* co     = ws + 851968;        // 64*2560
    float* selbuf = ws + 1015808;       // 64
    float* logits = ws + 1016832;       // 64*50000   (total ~16.9 MB)

    // ---- output layout (floats, concat in return order) ----
    float* out      = (float*)d_out;
    float* o_log    = out;                        // B*VOFF
    float* o_hid    = out + (size_t)B_ * VOFF_;   // 2*B*H
    float* o_attn   = o_hid + 2 * B_ * H_;        // B*S
    float* o_copy   = o_attn + B_ * S_;           // B*S
    float* o_sel    = o_copy + B_ * S_;           // B

    const float* h_old0 = last_h;               // layer 0 hidden
    const float* h_old1 = last_h + B_ * H_;     // layer 1 hidden (top)

    // 1. embedding gather/blend (-> emb, rnn_in[:, :E])
    embed_blend<<<(B_ * E_ + 255) / 256, 256, 0, stream>>>(tok, table, copybank, emb, rnn_in);

    // 2. attention query projections (q = h_top @ W^T), N=K=1024
    wmma_gemm_atb<<<32, 256, 0, stream>>>(h_old1, attn_W,  nullptr, qm, H_, H_);
    wmma_gemm_atb<<<32, 256, 0, stream>>>(h_old1, cattn_W, nullptr, qc, H_, H_);

    // 3. main attention (context -> ctx, rnn_in[:, E:]); copy attention (weights only)
    attn_softmax_ctx<<<B_, 256, 0, stream>>>(qm, mem, enc_len, 0, o_attn, ctx, rnn_in);
    attn_softmax_ctx<<<B_, 256, 0, stream>>>(qc, mem, enc_len, 1, o_copy, nullptr, nullptr);

    // 4. GRU layer 0: gi = rnn_in @ Wih0^T + bih0 ; gh = h_old0 @ Whh0^T + bhh0
    wmma_gemm_atb<<<96, 256, 0, stream>>>(rnn_in, Wih0, bih0, gi, 3 * H_, H_ + E_);
    wmma_gemm_atb<<<96, 256, 0, stream>>>(h_old0, Whh0, bhh0, gh, 3 * H_, H_);
    gru_gates<<<(B_ * H_ + 255) / 256, 256, 0, stream>>>(gi, gh, h_old0, h0, o_hid);

    // 5. GRU layer 1
    wmma_gemm_atb<<<96, 256, 0, stream>>>(h0,     Wih1, bih1, gi, 3 * H_, H_);
    wmma_gemm_atb<<<96, 256, 0, stream>>>(h_old1, Whh1, bhh1, gh, 3 * H_, H_);
    gru_gates<<<(B_ * H_ + 255) / 256, 256, 0, stream>>>(gi, gh, h_old1, h1, o_hid + B_ * H_);

    // 6. context_output = [h1; ctx; emb]; selector
    concat_ctxout<<<(B_ * (2 * H_ + E_) + 255) / 256, 256, 0, stream>>>(h1, ctx, emb, co);
    selector_k<<<B_, 256, 0, stream>>>(co, sel_w, selbuf, o_sel);

    // 7. big output projection: logits = co @ out_W^T + out_b  (64 x 50000 x 2560)
    wmma_gemm_atb<<<(V_ + 31) / 32, 256, 0, stream>>>(co, out_W, out_b, logits, V_, 2 * H_ + E_);

    // 8. softmax over V, blend with copy probs, log
    final_blend<<<B_, 256, 0, stream>>>(logits, selbuf, o_copy, o_log);
}
